// Attention_30391188586757
// MI455X (gfx1250) — matrix operand order repaired and checked by reading, not compiled, not run on hardware
//
#include <hip/hip_runtime.h>
#include <stdint.h>
#include <stddef.h>

// ---------------- constants ----------------
constexpr int kB   = 2;
constexpr int kS   = 2048;
constexpr int kHID = 2560;
constexpr int kH   = 32;
constexpr int kKV  = 8;
constexpr int kD   = 128;
constexpr int kG   = kH / kKV;       // 4
constexpr int kM   = kB * kS;        // 4096 tokens

#define DEVI __device__ __forceinline__

typedef __bf16 bf16_t;
typedef __attribute__((ext_vector_type(16))) bf16_t          v16bf;
typedef __attribute__((ext_vector_type(8)))  float           v8f;
typedef __attribute__((ext_vector_type(8)))  unsigned short  u16x8;
typedef __attribute__((ext_vector_type(4)))  unsigned int    u32x4;
typedef __attribute__((ext_vector_type(4)))  int             i32x4;
typedef __attribute__((ext_vector_type(8)))  int             i32x8;

union FragAB { v16bf v; u16x8 h[2]; unsigned short s[16]; };

DEVI unsigned short f2bf(float x) {
  union { float f; unsigned int u; } c; c.f = x;
  unsigned int u = c.u;
  unsigned int r = u + 0x7FFFu + ((u >> 16) & 1u);   // round-to-nearest-even
  return (unsigned short)(r >> 16);
}

DEVI v8f v8zero() { v8f r; for (int i = 0; i < 8; ++i) r[i] = 0.0f; return r; }

// 16-byte async copy global -> LDS (no VGPR round-trip, tracked by ASYNCcnt)
DEVI void async_cp16(unsigned short* ldst, const unsigned short* gsrc) {
  const unsigned lo = (unsigned)(uintptr_t)ldst;            // LDS byte address
  const unsigned long long ga = (unsigned long long)(uintptr_t)gsrc;
  asm volatile("global_load_async_to_lds_b128 %0, %1, off"
               :: "v"(lo), "v"(ga) : "memory");
}
DEVI void wait_async0() { asm volatile("s_wait_asynccnt 0" ::: "memory"); }

// ---------------- fp32 -> bf16 conversion ----------------
__global__ void cvt_f32_bf16(const float* __restrict__ in,
                             unsigned short* __restrict__ out, int n) {
  int i = blockIdx.x * blockDim.x + threadIdx.x;
  if (i < n) out[i] = f2bf(in[i]);
}

// ---------------- NT GEMM: C[M,N] = A[M,K] * B[N,K]^T  (bf16 in, f32 out) ----
constexpr int BM = 128, BN = 128, BK = 32, LDT = BK + 8;   // LDS row stride (shorts)

__global__ __launch_bounds__(256) void gemm_nt_bf16(
    const unsigned short* __restrict__ A, const unsigned short* __restrict__ B,
    float* __restrict__ C, int M, int N, int K) {
  __shared__ unsigned short As[2][BM * LDT];
  __shared__ unsigned short Bs[2][BN * LDT];

  const int t    = threadIdx.x;
  const int lane = t & 31, wid = t >> 5;
  const int wm   = wid >> 1, wn = wid & 1;          // 4x2 wave grid
  const int bm   = blockIdx.y * BM, bn = blockIdx.x * BN;
  const int l16  = lane & 15, g = lane >> 4;

  v8f acc[2][4];
  for (int mi = 0; mi < 2; ++mi)
    for (int ni = 0; ni < 4; ++ni) acc[mi][ni] = v8zero();

  // each thread stages 2x 16B chunks per tile per matrix, all async->LDS
  const int lr = t >> 2, lc8 = (t & 3) * 8;

  auto load_tiles = [&](int buf, int k0) {
    async_cp16(&As[buf][(lr)      * LDT + lc8], A + (size_t)(bm + lr)      * K + k0 + lc8);
    async_cp16(&As[buf][(lr + 64) * LDT + lc8], A + (size_t)(bm + lr + 64) * K + k0 + lc8);
    async_cp16(&Bs[buf][(lr)      * LDT + lc8], B + (size_t)(bn + lr)      * K + k0 + lc8);
    async_cp16(&Bs[buf][(lr + 64) * LDT + lc8], B + (size_t)(bn + lr + 64) * K + k0 + lc8);
  };

  load_tiles(0, 0);
  wait_async0();
  __syncthreads();

  const int nk = K / BK;
  for (int kt = 0; kt < nk; ++kt) {
    const int cur = kt & 1;
    if (kt + 1 < nk) load_tiles(cur ^ 1, (kt + 1) * BK);   // async, in flight under WMMAs
    if (kt + 2 < nk) {   // global_prefetch_b8 of the K+2 tile
      __builtin_prefetch(A + (size_t)(bm + lr) * K + (kt + 2) * BK + lc8, 0, 0);
      __builtin_prefetch(B + (size_t)(bn + lr) * K + (kt + 2) * BK + lc8, 0, 0);
    }

    const unsigned short* as = As[cur];
    const unsigned short* bs = Bs[cur];

    FragAB af[2], bfr[4];
    for (int mi = 0; mi < 2; ++mi) {
      const int row = wm * 32 + mi * 16 + l16;
      af[mi].h[0] = *(const u16x8*)&as[row * LDT + 8 * g];
      af[mi].h[1] = *(const u16x8*)&as[row * LDT + 16 + 8 * g];
    }
    for (int ni = 0; ni < 4; ++ni) {
      const int col = wn * 64 + ni * 16 + l16;
      bfr[ni].h[0] = *(const u16x8*)&bs[col * LDT + 8 * g];
      bfr[ni].h[1] = *(const u16x8*)&bs[col * LDT + 16 + 8 * g];
    }
    for (int mi = 0; mi < 2; ++mi)
      for (int ni = 0; ni < 4; ++ni)
        acc[mi][ni] = __builtin_amdgcn_wmma_f32_16x16x32_bf16(
            false, af[mi].v, false, bfr[ni].v, (short)0, acc[mi][ni], false, false);

    wait_async0();       // next tile landed
    __syncthreads();
  }

  for (int mi = 0; mi < 2; ++mi)
    for (int ni = 0; ni < 4; ++ni)
      for (int v = 0; v < 8; ++v) {
        const int row = bm + wm * 32 + mi * 16 + g * 8 + v;
        const int col = bn + wn * 64 + ni * 16 + l16;
        C[(size_t)row * N + col] = acc[mi][ni][v];
      }
}

// ---------------- RMSNorm + RoPE (Q) : [m, H*D] f32 -> (b,H,s,D) bf16 -------
__global__ __launch_bounds__(kD) void rmsnorm_rope_q(
    const float* __restrict__ X, const float* __restrict__ cosb,
    const float* __restrict__ sinb, const float* __restrict__ gamma,
    unsigned short* __restrict__ out) {
  const int m = blockIdx.x / kH, h = blockIdx.x % kH, d = threadIdx.x;
  const float x = X[(size_t)blockIdx.x * kD + d];
  __shared__ float sred[kD], sxn[kD];
  sred[d] = x * x;
  __syncthreads();
  for (int s = 64; s; s >>= 1) { if (d < s) sred[d] += sred[d + s]; __syncthreads(); }
  const float inv = rsqrtf(sred[0] / (float)kD + 1e-6f);
  const float xn  = x * inv * gamma[d];
  sxn[d] = xn;
  __syncthreads();
  const float rot = (d < 64) ? -sxn[d + 64] : sxn[d - 64];
  const int b = m / kS, sidx = m % kS;
  const size_t co = (size_t)m * kD + d;
  const float r = xn * cosb[co] + rot * sinb[co];
  out[(((size_t)(b * kH + h)) * kS + sidx) * kD + d] = f2bf(r);
}

// ---------------- RMSNorm + RoPE (K) : also writes fp32 new_k ---------------
__global__ __launch_bounds__(kD) void rmsnorm_rope_k(
    const float* __restrict__ X, const float* __restrict__ cosb,
    const float* __restrict__ sinb, const float* __restrict__ gamma,
    float* __restrict__ kout, unsigned short* __restrict__ kbf) {
  const int m = blockIdx.x / kKV, h = blockIdx.x % kKV, d = threadIdx.x;
  const float x = X[(size_t)blockIdx.x * kD + d];
  __shared__ float sred[kD], sxn[kD];
  sred[d] = x * x;
  __syncthreads();
  for (int s = 64; s; s >>= 1) { if (d < s) sred[d] += sred[d + s]; __syncthreads(); }
  const float inv = rsqrtf(sred[0] / (float)kD + 1e-6f);
  const float xn  = x * inv * gamma[d];
  sxn[d] = xn;
  __syncthreads();
  const float rot = (d < 64) ? -sxn[d + 64] : sxn[d - 64];
  const int b = m / kS, sidx = m % kS;
  const size_t co = (size_t)m * kD + d;
  const float r = xn * cosb[co] + rot * sinb[co];
  const size_t o = (((size_t)(b * kKV + h)) * kS + sidx) * kD + d;
  kout[o] = r;
  kbf[o]  = f2bf(r);
}

// ---------------- V relayout: [m, KV*D] f32 -> (b,KV,s,D) f32 + bf16 --------
__global__ __launch_bounds__(kD) void v_relayout(
    const float* __restrict__ X, float* __restrict__ vout,
    unsigned short* __restrict__ vbf) {
  const int m = blockIdx.x / kKV, h = blockIdx.x % kKV, d = threadIdx.x;
  const float x = X[(size_t)blockIdx.x * kD + d];
  const int b = m / kS, sidx = m % kS;
  const size_t o = (((size_t)(b * kKV + h)) * kS + sidx) * kD + d;
  vout[o] = x;
  vbf[o]  = f2bf(x);
}

// ---------------- flash attention: causal GQA, bf16 WMMA, f32 softmax -------
constexpr int FT = 64;   // q-rows and kv-cols per tile

__global__ __launch_bounds__(128) void flash_attn(
    const unsigned short* __restrict__ Q, const unsigned short* __restrict__ Kb,
    const unsigned short* __restrict__ Vb, unsigned short* __restrict__ O) {
  __shared__ unsigned short qs[FT][kD + 8];        // 17408 B (TDM-filled, padded)
  __shared__ unsigned short ks[FT][kD + 8];        // 17408 B (async-filled)
  __shared__ unsigned short vts[kD][FT + 8];       // 18432 B (V transposed)
  __shared__ unsigned short ps[4][16][FT + 8];     //  9216 B (per-wave P)

  const int t = threadIdx.x, lane = t & 31, w = t >> 5;
  const int l16 = lane & 15, g = lane >> 4;
  const int bh = blockIdx.y, b = bh / kH, h = bh % kH, kv = h / kG;
  const int q0 = blockIdx.x * FT;

  const size_t qbase = (((size_t)(b * kH + h)) * kS + q0) * kD;

  // ---- Q tile via Tensor Data Mover: 2D tile 128 x 64 elems (2B each),
  // ---- LDS padding 4 DWORDs every 64 DWORDs reproduces the kD+8 row stride.
  if (w == 0) {
    const unsigned long long ga = (unsigned long long)(uintptr_t)(Q + qbase);
    const unsigned lds_q = (unsigned)(uintptr_t)&qs[0][0];
    u32x4 g0 = { 1u,                                  // count=1, user mode
                 lds_q,                               // lds_addr
                 (unsigned)(ga & 0xFFFFFFFFu),        // global_addr[31:0]
                 (unsigned)((ga >> 32) & 0x01FFFFFFu) | (2u << 30) }; // type=2
    i32x8 g1 = { (int)((1u << 16) | (1u << 20) | (5u << 22) | (3u << 25)),
                 //  data_size=2B | pad_enable | pad_interval=64DW | pad_amount=4DW
                 (int)(128u << 16),                   // tensor_dim0 = 128
                 (int)(64u  << 16),                   // tensor_dim1 = 64
                 (int)(128u << 16),                   // tile_dim0   = 128
                 64,                                  // tile_dim1   = 64
                 128,                                 // tensor_dim0_stride = 128
                 0, 0 };
    i32x4 g2 = { 0, 0, 0, 0 };                        // 2D tensor: groups 2/3 unused
    i32x4 g3 = { 0, 0, 0, 0 };
    i32x8 g4 = { 0, 0, 0, 0, 0, 0, 0, 0 };           // 6-arg toolchain: extra group
    __builtin_amdgcn_tensor_load_to_lds(g0, g1, g2, g3, g4, 0);
  }
  __builtin_amdgcn_s_wait_tensorcnt(0);              // s_wait_tensorcnt 0

  v8f o_acc[8];
  for (int i = 0; i < 8; ++i) o_acc[i] = v8zero();
  float mrow[8], lrow[8];
  for (int i = 0; i < 8; ++i) { mrow[i] = -1e30f; lrow[i] = 0.0f; }

  const float scale = 0.08838834764831845f;          // 1/sqrt(128)
  const size_t kbase = ((size_t)(b * kKV + kv)) * kS * kD;

  for (int kt = 0; kt <= blockIdx.x; ++kt) {
    const int l0 = kt * FT;
    __syncthreads();                                 // qs ready / ks,vts reusable
    for (int e = t; e < FT * kD / 8; e += 128) {     // K tile: async global->LDS
      const int row = e / 16, c8 = (e % 16) * 8;
      async_cp16(&ks[row][c8],
                 Kb + kbase + (size_t)(l0 + row) * kD + c8);
    }
    for (int e = t; e < FT * kD; e += 128) {         // V tile transposed (VGPR path)
      const int l = e >> 7, d = e & 127;
      vts[d][l] = Vb[kbase + (size_t)(l0 + l) * kD + d];
    }
    wait_async0();
    __syncthreads();

    // S = Q K^T  (this wave's 16 q-rows x 64 kv-cols)
    v8f sacc[4];
    for (int ni = 0; ni < 4; ++ni) sacc[ni] = v8zero();
    const int arow = w * 16 + l16;
    for (int kk = 0; kk < 4; ++kk) {
      FragAB af;
      af.h[0] = *(const u16x8*)&qs[arow][kk * 32 + 8 * g];
      af.h[1] = *(const u16x8*)&qs[arow][kk * 32 + 16 + 8 * g];
      for (int ni = 0; ni < 4; ++ni) {
        FragAB bfr;
        const int col = ni * 16 + l16;
        bfr.h[0] = *(const u16x8*)&ks[col][kk * 32 + 8 * g];
        bfr.h[1] = *(const u16x8*)&ks[col][kk * 32 + 16 + 8 * g];
        sacc[ni] = __builtin_amdgcn_wmma_f32_16x16x32_bf16(
            false, af.v, false, bfr.v, (short)0, sacc[ni], false, false);
      }
    }

    // scale + causal mask + online softmax (per row: v + 8*g)
    for (int v = 0; v < 8; ++v) {
      const int qr = q0 + w * 16 + g * 8 + v;
      float mx = -1e30f;
      for (int ni = 0; ni < 4; ++ni) {
        float s = sacc[ni][v] * scale;
        const int lc = l0 + ni * 16 + l16;
        if (lc > qr) s = -1e30f;
        sacc[ni][v] = s;
        mx = fmaxf(mx, s);
      }
      for (int i = 1; i < 16; i <<= 1) mx = fmaxf(mx, __shfl_xor(mx, i, 32));
      const float mnew  = fmaxf(mrow[v], mx);
      const float alpha = __expf(mrow[v] - mnew);
      float rs = 0.0f;
      for (int ni = 0; ni < 4; ++ni) {
        const float p = __expf(sacc[ni][v] - mnew);
        sacc[ni][v] = p;
        rs += p;
      }
      for (int i = 1; i < 16; i <<= 1) rs += __shfl_xor(rs, i, 32);
      lrow[v] = lrow[v] * alpha + rs;
      mrow[v] = mnew;
      for (int nf = 0; nf < 8; ++nf) o_acc[nf][v] *= alpha;
    }

    // stage P (bf16) in this wave's private LDS slab
    for (int v = 0; v < 8; ++v) {
      const int pr = g * 8 + v;
      for (int ni = 0; ni < 4; ++ni)
        ps[w][pr][ni * 16 + l16] = f2bf(sacc[ni][v]);
    }
    asm volatile("s_wait_dscnt 0" ::: "memory");     // P visible to whole wave

    // O += P * V   (K-dim = 64)
    for (int kk = 0; kk < 2; ++kk) {
      FragAB af;
      af.h[0] = *(const u16x8*)&ps[w][l16][kk * 32 + 8 * g];
      af.h[1] = *(const u16x8*)&ps[w][l16][kk * 32 + 16 + 8 * g];
      for (int nf = 0; nf < 8; ++nf) {
        FragAB bfr;
        const int d = nf * 16 + l16;
        bfr.h[0] = *(const u16x8*)&vts[d][kk * 32 + 8 * g];
        bfr.h[1] = *(const u16x8*)&vts[d][kk * 32 + 16 + 8 * g];
        o_acc[nf] = __builtin_amdgcn_wmma_f32_16x16x32_bf16(
            false, af.v, false, bfr.v, (short)0, o_acc[nf], false, false);
      }
    }
  }

  // epilogue: O / l  -> attn_bf[m, h*D + d]
  for (int v = 0; v < 8; ++v) {
    const int qr = q0 + w * 16 + g * 8 + v;
    const float inv = 1.0f / lrow[v];
    const size_t rowo = ((size_t)(b * kS + qr)) * (size_t)(kH * kD) + (size_t)h * kD;
    for (int nf = 0; nf < 8; ++nf)
      O[rowo + nf * 16 + l16] = f2bf(o_acc[nf][v] * inv);
  }
}

// ---------------- host-side orchestration ----------------
extern "C" void kernel_launch(void* const* d_in, const int* in_sizes, int n_in,
                              void* d_out, int out_size, void* d_ws, size_t ws_size,
                              hipStream_t stream) {
  const float* hidden = (const float*)d_in[0];
  const float* cosb   = (const float*)d_in[1];
  const float* sinb   = (const float*)d_in[2];
  const float* wq     = (const float*)d_in[3];
  const float* wk     = (const float*)d_in[4];
  const float* wv     = (const float*)d_in[5];
  const float* wo     = (const float*)d_in[6];
  const float* qg     = (const float*)d_in[7];
  const float* kg     = (const float*)d_in[8];

  float* out   = (float*)d_out;
  float* k_out = out + (size_t)kM * kHID;
  float* v_out = k_out + (size_t)kB * kKV * kS * kD;

  char* p = (char*)d_ws;
  auto alloc = [&](size_t bytes) -> void* {
    void* r = (void*)p;
    p += (bytes + 255) & ~(size_t)255;
    return r;
  };
  unsigned short* hbf  = (unsigned short*)alloc((size_t)kM * kHID * 2);
  unsigned short* wqb  = (unsigned short*)alloc((size_t)kH * kD * kHID * 2);
  unsigned short* wkb  = (unsigned short*)alloc((size_t)kKV * kD * kHID * 2);
  unsigned short* wvb  = (unsigned short*)alloc((size_t)kKV * kD * kHID * 2);
  unsigned short* wob  = (unsigned short*)alloc((size_t)kHID * kH * kD * 2);
  float*          qraw = (float*)alloc((size_t)kM * kH * kD * 4);
  float*          kraw = (float*)alloc((size_t)kM * kKV * kD * 4);
  float*          vraw = (float*)alloc((size_t)kM * kKV * kD * 4);
  unsigned short* qbf  = (unsigned short*)alloc((size_t)kM * kH * kD * 2);
  unsigned short* kbf  = (unsigned short*)alloc((size_t)kM * kKV * kD * 2);
  unsigned short* vbf  = (unsigned short*)alloc((size_t)kM * kKV * kD * 2);
  unsigned short* attn = (unsigned short*)qraw;   // reuse: qraw dead after rope_q

  auto cvt = [&](const float* src, unsigned short* dst, int n) {
    cvt_f32_bf16<<<(n + 255) / 256, 256, 0, stream>>>(src, dst, n);
  };
  cvt(hidden, hbf, kM * kHID);
  cvt(wq, wqb, kH * kD * kHID);
  cvt(wk, wkb, kKV * kD * kHID);
  cvt(wv, wvb, kKV * kD * kHID);
  cvt(wo, wob, kHID * kH * kD);

  gemm_nt_bf16<<<dim3(kH * kD / BN, kM / BM), 256, 0, stream>>>(
      hbf, wqb, qraw, kM, kH * kD, kHID);
  gemm_nt_bf16<<<dim3(kKV * kD / BN, kM / BM), 256, 0, stream>>>(
      hbf, wkb, kraw, kM, kKV * kD, kHID);
  gemm_nt_bf16<<<dim3(kKV * kD / BN, kM / BM), 256, 0, stream>>>(
      hbf, wvb, vraw, kM, kKV * kD, kHID);

  rmsnorm_rope_q<<<kM * kH, kD, 0, stream>>>(qraw, cosb, sinb, qg, qbf);
  rmsnorm_rope_k<<<kM * kKV, kD, 0, stream>>>(kraw, cosb, sinb, kg, k_out, kbf);
  v_relayout<<<kM * kKV, kD, 0, stream>>>(vraw, v_out, vbf);

  flash_attn<<<dim3(kS / FT, kB * kH), 128, 0, stream>>>(qbf, kbf, vbf, attn);

  gemm_nt_bf16<<<dim3(kHID / BN, kM / BM), 256, 0, stream>>>(
      attn, wob, out, kM, kHID, kH * kD);
}
